// AttentionEncoder_27565100106033
// MI455X (gfx1250) — matrix operand order repaired and checked by reading, not compiled, not run on hardware
//
#include <hip/hip_runtime.h>

// ---------------------------------------------------------------------------
// Problem constants (from the reference)
// ---------------------------------------------------------------------------
#define N_NODES   50000
#define N_EDGES   800000
#define N_ETOT    (N_EDGES + N_NODES)   // with self loops
#define N_GRAPHS  256
#define N_IN      300
#define HID_ATT   128
#define NHID      300
#define NOUT      768
#define SLOPE     0.2f

typedef __attribute__((ext_vector_type(16))) _Float16 v16h;
typedef __attribute__((ext_vector_type(8)))  _Float16 v8h;
typedef __attribute__((ext_vector_type(8)))  float    v8f;

static inline int ceil32(int x) { return (x + 31) & ~31; }

// ---------------------------------------------------------------------------
// Pre-convert activations: fp32 M x K  ->  f16 M x Kpad (zero padded),
// optional fused relu.  Makes GEMM A-fragments direct aligned global loads.
// ---------------------------------------------------------------------------
__global__ void convert_a_kernel(const float* __restrict__ A,
                                 _Float16* __restrict__ Ah,
                                 int K, int Kpad, int relu) {
  int k   = blockIdx.x * blockDim.x + threadIdx.x;
  int row = blockIdx.y;
  if (k >= Kpad) return;
  float v = (k < K) ? A[(size_t)row * K + k] : 0.f;
  if (relu) v = fmaxf(v, 0.f);
  Ah[(size_t)row * Kpad + k] = (_Float16)v;
}

// ---------------------------------------------------------------------------
// Pre-convert weights: fp32 K x N -> f16 transposed Npad x Kpad (zero padded).
// B-fragments become 16 contiguous halves per lane (one 32B load).
// ---------------------------------------------------------------------------
__global__ void convert_w_kernel(const float* __restrict__ W,
                                 _Float16* __restrict__ Wt,
                                 int K, int N, int Kpad) {
  int k = blockIdx.x * blockDim.x + threadIdx.x;
  int n = blockIdx.y;
  if (k >= Kpad) return;
  float v = (k < K && n < N) ? W[(size_t)k * N + n] : 0.f;
  Wt[(size_t)n * Kpad + k] = (_Float16)v;
}

// ---------------------------------------------------------------------------
// WMMA GEMM: one wave computes a 16x64 tile of C = Ah @ Wt^T + bias.
// Ah: f16 M x Kpad (row major, padded).  Wt: f16 Npad x Kpad (W transposed).
// Inner loop: 2 aligned A loads + 4x1 aligned B loads + 4 v_wmma, no LDS,
// no barriers, no predication (padding guarantees validity).
// ---------------------------------------------------------------------------
__global__ void wmma_gemm_kernel(const _Float16* __restrict__ Ah,
                                 const _Float16* __restrict__ Wt,
                                 const float* __restrict__ bias,
                                 float* __restrict__ Cmat,
                                 int M, int N, int Kpad, int nColTiles,
                                 int reluOut) {
  const int gw   = (int)(((long long)blockIdx.x * blockDim.x + threadIdx.x) >> 5);
  const int lane = threadIdx.x & 31;
  const int rowTiles = M >> 4;
  if (gw >= rowTiles * nColTiles) return;
  const int row0 = (gw / nColTiles) << 4;
  const int col0 = (gw % nColTiles) << 6;

  const int lo16 = (lane < 16);
  const int arow = row0 + (lane & 15);
  const int kA   = lo16 ? 0 : 16;    // A-fragment K sub-offset
  const int kB   = lo16 ? 0 : 16;    // B-fragment K sub-offset
  const int bcol = col0 + (lane & 15);

  const _Float16* ap = Ah + (size_t)arow * Kpad + kA;
  const _Float16* bp0 = Wt + (size_t)(bcol +  0) * Kpad + kB;
  const _Float16* bp1 = Wt + (size_t)(bcol + 16) * Kpad + kB;
  const _Float16* bp2 = Wt + (size_t)(bcol + 32) * Kpad + kB;
  const _Float16* bp3 = Wt + (size_t)(bcol + 48) * Kpad + kB;

  v8f acc0 = {}, acc1 = {}, acc2 = {}, acc3 = {};
  for (int k0 = 0; k0 < Kpad; k0 += 32) {
    v8h alo = *(const v8h*)(ap + k0);
    v8h ahi = *(const v8h*)(ap + k0 + 8);
    v16h af = __builtin_shufflevector(alo, ahi,
        0, 1, 2, 3, 4, 5, 6, 7, 8, 9, 10, 11, 12, 13, 14, 15);
    v16h b0 = *(const v16h*)(bp0 + k0);
    v16h b1 = *(const v16h*)(bp1 + k0);
    v16h b2 = *(const v16h*)(bp2 + k0);
    v16h b3 = *(const v16h*)(bp3 + k0);
    acc0 = __builtin_amdgcn_wmma_f32_16x16x32_f16(false, af, false, b0, (short)0, acc0, false, false);
    acc1 = __builtin_amdgcn_wmma_f32_16x16x32_f16(false, af, false, b1, (short)0, acc1, false, false);
    acc2 = __builtin_amdgcn_wmma_f32_16x16x32_f16(false, af, false, b2, (short)0, acc2, false, false);
    acc3 = __builtin_amdgcn_wmma_f32_16x16x32_f16(false, af, false, b3, (short)0, acc3, false, false);
  }

  // C/D layout: VGPR r -> row r (lanes 0..15) / row r+8 (lanes 16..31)
  const int rbase = row0 + (lo16 ? 0 : 8);
  v8f accs[4] = {acc0, acc1, acc2, acc3};
#pragma unroll
  for (int j = 0; j < 4; ++j) {
    const int col = bcol + 16 * j;
    if (col < N) {
      const float bv = bias[col];
#pragma unroll
      for (int r = 0; r < 8; ++r) {
        float v = accs[j][r] + bv;
        if (reluOut) v = fmaxf(v, 0.f);
        Cmat[(size_t)(rbase + r) * N + col] = v;
      }
    }
  }
}

// ---------------------------------------------------------------------------
// Float atomic max via signed/unsigned int atomics (IEEE ordering trick).
// ---------------------------------------------------------------------------
__device__ __forceinline__ void atomicMaxFloat(float* addr, float val) {
  if (val >= 0.f) atomicMax((int*)addr, __float_as_int(val));
  else            atomicMin((unsigned int*)addr, (unsigned int)__float_as_int(val));
}

__device__ __forceinline__ void edge_src_dst(const int* __restrict__ ei, int e,
                                             int& s, int& d) {
  if (e < N_EDGES) { s = ei[e]; d = ei[N_EDGES + e]; }
  else             { s = e - N_EDGES; d = s; }      // self loop
}

// ---------------------------------------------------------------------------
// Per-layer init: out = bias broadcast (2D grid, no modulo), mx/denom reset.
// ---------------------------------------------------------------------------
__global__ void init_out_kernel(float* __restrict__ out,
                                const float* __restrict__ bias, int C) {
  int c = blockIdx.x * blockDim.x + threadIdx.x;
  if (c < C) out[(size_t)blockIdx.y * C + c] = bias[c];
}

__global__ void init_seg_kernel(float* __restrict__ mx, float* __restrict__ denom) {
  int i = blockIdx.x * blockDim.x + threadIdx.x;
  if (i < N_NODES) {
    mx[i]    = __int_as_float((int)0xFF800000u);   // -inf
    denom[i] = 0.f;
  }
}

// ---------------------------------------------------------------------------
// Wave per edge: logit = att . leaky_relu(xl[src] + xr[dst]); segment max.
// Channels processed as float4 (C % 4 == 0 for 128 and 300).
// ---------------------------------------------------------------------------
__global__ void edge_logits_kernel(const float4* __restrict__ xl4,
                                   const float4* __restrict__ xr4,
                                   const int* __restrict__ ei,
                                   const float4* __restrict__ att4,
                                   float* __restrict__ logits,
                                   float* __restrict__ mx, int C4) {
  int e    = (int)(((long long)blockIdx.x * blockDim.x + threadIdx.x) >> 5);
  int lane = threadIdx.x & 31;
  if (e >= N_ETOT) return;
  int s, d;
  edge_src_dst(ei, e, s, d);
  float part = 0.f;
  for (int c = lane; c < C4; c += 32) {
    float4 a = xl4[(size_t)s * C4 + c];
    float4 b = xr4[(size_t)d * C4 + c];
    float4 t = att4[c];
    float m;
    m = a.x + b.x; part += ((m > 0.f) ? m : SLOPE * m) * t.x;
    m = a.y + b.y; part += ((m > 0.f) ? m : SLOPE * m) * t.y;
    m = a.z + b.z; part += ((m > 0.f) ? m : SLOPE * m) * t.z;
    m = a.w + b.w; part += ((m > 0.f) ? m : SLOPE * m) * t.w;
  }
  for (int off = 16; off > 0; off >>= 1) part += __shfl_xor(part, off, 32);
  if (lane == 0) {
    logits[e] = part;
    atomicMaxFloat(mx + d, part);
  }
}

// ---------------------------------------------------------------------------
// a = exp(logit - mx[dst]); denom[dst] += a.
// ---------------------------------------------------------------------------
__global__ void edge_exp_kernel(const float* __restrict__ logits,
                                const float* __restrict__ mx,
                                const int* __restrict__ ei,
                                float* __restrict__ aex,
                                float* __restrict__ denom) {
  int e = blockIdx.x * blockDim.x + threadIdx.x;
  if (e >= N_ETOT) return;
  int d = (e < N_EDGES) ? ei[N_EDGES + e] : (e - N_EDGES);
  float v = __expf(logits[e] - mx[d]);
  aex[e] = v;
  atomicAdd(denom + d, v);
}

// ---------------------------------------------------------------------------
// Wave per edge: out[dst] += (a/denom[dst]) * xl[src]  (float4 loads).
// ---------------------------------------------------------------------------
__global__ void edge_scatter_kernel(const float4* __restrict__ xl4,
                                    const float* __restrict__ aex,
                                    const float* __restrict__ denom,
                                    const int* __restrict__ ei,
                                    float* __restrict__ out, int C4) {
  int e    = (int)(((long long)blockIdx.x * blockDim.x + threadIdx.x) >> 5);
  int lane = threadIdx.x & 31;
  if (e >= N_ETOT) return;
  int s, d;
  edge_src_dst(ei, e, s, d);
  float alpha = aex[e] / denom[d];
  for (int c = lane; c < C4; c += 32) {
    float4 v = xl4[(size_t)s * C4 + c];
    float* o = out + ((size_t)d * C4 + c) * 4;
    atomicAdd(o + 0, alpha * v.x);
    atomicAdd(o + 1, alpha * v.y);
    atomicAdd(o + 2, alpha * v.z);
    atomicAdd(o + 3, alpha * v.w);
  }
}

// ---------------------------------------------------------------------------
// Pooling: zero, scatter (wave per node, fused relu, float4), divide.
// ---------------------------------------------------------------------------
__global__ void zero_kernel(float* __restrict__ p, int n) {
  int i = blockIdx.x * blockDim.x + threadIdx.x;
  if (i < n) p[i] = 0.f;
}

__global__ void pool_scatter_kernel(const float4* __restrict__ h4,
                                    const int* __restrict__ batch,
                                    float* __restrict__ s,
                                    float* __restrict__ cnt, int C4) {
  int n    = (int)(((long long)blockIdx.x * blockDim.x + threadIdx.x) >> 5);
  int lane = threadIdx.x & 31;
  if (n >= N_NODES) return;
  int b = batch[n];
  for (int c = lane; c < C4; c += 32) {
    float4 v = h4[(size_t)n * C4 + c];
    float* o = s + ((size_t)b * C4 + c) * 4;
    atomicAdd(o + 0, fmaxf(v.x, 0.f));     // encoder relu before pool
    atomicAdd(o + 1, fmaxf(v.y, 0.f));
    atomicAdd(o + 2, fmaxf(v.z, 0.f));
    atomicAdd(o + 3, fmaxf(v.w, 0.f));
  }
  if (lane == 0) atomicAdd(cnt + b, 1.f);
}

__global__ void pool_div_kernel(const float* __restrict__ s,
                                const float* __restrict__ cnt,
                                float* __restrict__ g, int C) {
  int i = blockIdx.x * blockDim.x + threadIdx.x;
  if (i < N_GRAPHS * C) g[i] = s[i] / fmaxf(cnt[i / C], 1.f);
}

// ---------------------------------------------------------------------------
// Host-side launchers
// ---------------------------------------------------------------------------
static inline void launch_convert_a(const float* A, _Float16* Ah, int M, int K,
                                    int Kpad, int relu, hipStream_t stream) {
  dim3 grid((Kpad + 255) / 256, M);
  convert_a_kernel<<<grid, 256, 0, stream>>>(A, Ah, K, Kpad, relu);
}

static inline void launch_convert_w(const float* W, _Float16* Wt, int K, int N,
                                    int Kpad, int Npad, hipStream_t stream) {
  dim3 grid((Kpad + 255) / 256, Npad);
  convert_w_kernel<<<grid, 256, 0, stream>>>(W, Wt, K, N, Kpad);
}

static inline void launch_gemm(const _Float16* Ah, const _Float16* Wt,
                               const float* bias, float* Cmat,
                               int M, int N, int Kpad, int reluOut,
                               hipStream_t stream) {
  int nColTiles = (N + 63) / 64;
  long long waves = (long long)(M / 16) * nColTiles;
  int blocks = (int)((waves + 7) / 8);         // 8 waves / 256-thread block
  wmma_gemm_kernel<<<blocks, 256, 0, stream>>>(Ah, Wt, bias, Cmat, M, N, Kpad,
                                               nColTiles, reluOut);
}

extern "C" void kernel_launch(void* const* d_in, const int* in_sizes, int n_in,
                              void* d_out, int out_size, void* d_ws, size_t ws_size,
                              hipStream_t stream) {
  (void)in_sizes; (void)n_in; (void)out_size; (void)ws_size;

  const float* x     = (const float*)d_in[0];
  const int*   ei    = (const int*)d_in[1];
  const int*   batch = (const int*)d_in[2];

  // ---- workspace carve (floats, 64-float = 256B aligned chunks) ----
  float* ws = (float*)d_ws;
  size_t off = 0;
  auto carve = [&](size_t n) { float* p = ws + off; off += (n + 63) & ~(size_t)63; return p; };
  float*    xl     = carve((size_t)N_NODES * NHID);
  float*    xr     = carve((size_t)N_NODES * NHID);
  float*    h0     = carve((size_t)N_NODES * NHID);
  float*    h1     = carve((size_t)N_NODES * NHID);
  _Float16* hh     = (_Float16*)carve((size_t)N_NODES * 320 / 2);  // f16 activations
  _Float16* wtA    = (_Float16*)carve((size_t)768 * 320 / 2);      // f16 W^T (padded)
  _Float16* wtB    = (_Float16*)carve((size_t)768 * 320 / 2);
  _Float16* gh     = (_Float16*)carve((size_t)N_GRAPHS * 320 / 2);
  float*    logits = carve(N_ETOT);
  float*    aex    = carve(N_ETOT);
  float*    mx     = carve(N_NODES);
  float*    denom  = carve(N_NODES);
  float*    pools  = carve((size_t)N_GRAPHS * NHID);
  float*    cnt    = carve(N_GRAPHS);
  float*    gbuf   = carve((size_t)N_GRAPHS * NHID);
  float*    fc1o   = carve((size_t)N_GRAPHS * NHID);

  const int Cin[4]  = {N_IN, HID_ATT, HID_ATT, HID_ATT};
  const int Cout[4] = {HID_ATT, HID_ATT, HID_ATT, NHID};

  const int edgeWaveBlocks = (N_ETOT + 7) / 8;
  const int edgeThrBlocks  = (N_ETOT + 255) / 256;

  const float* h = x;
  float* hout = h0;
  for (int l = 0; l < 4; ++l) {
    const float* Wl   = (const float*)d_in[3 + 6 * l + 0];
    const float* bl   = (const float*)d_in[3 + 6 * l + 1];
    const float* Wr   = (const float*)d_in[3 + 6 * l + 2];
    const float* br   = (const float*)d_in[3 + 6 * l + 3];
    const float* att  = (const float*)d_in[3 + 6 * l + 4];
    const float* bias = (const float*)d_in[3 + 6 * l + 5];
    const int ci = Cin[l], co = Cout[l];
    const int Kpad = ceil32(ci);
    const int Npad = ((co + 63) / 64) * 64;

    // f16 staging (inter-layer relu fused into activation convert)
    launch_convert_a(h, hh, N_NODES, ci, Kpad, (l > 0) ? 1 : 0, stream);
    launch_convert_w(Wl, wtA, ci, co, Kpad, Npad, stream);
    launch_convert_w(Wr, wtB, ci, co, Kpad, Npad, stream);

    launch_gemm(hh, wtA, bl, xl, N_NODES, co, Kpad, 0, stream);
    launch_gemm(hh, wtB, br, xr, N_NODES, co, Kpad, 0, stream);

    dim3 initGrid((co + 255) / 256, N_NODES);
    init_out_kernel<<<initGrid, 256, 0, stream>>>(hout, bias, co);
    init_seg_kernel<<<(N_NODES + 255) / 256, 256, 0, stream>>>(mx, denom);

    const int C4 = co / 4;
    edge_logits_kernel<<<edgeWaveBlocks, 256, 0, stream>>>(
        (const float4*)xl, (const float4*)xr, ei, (const float4*)att, logits, mx, C4);
    edge_exp_kernel<<<edgeThrBlocks, 256, 0, stream>>>(logits, mx, ei, aex, denom);
    edge_scatter_kernel<<<edgeWaveBlocks, 256, 0, stream>>>(
        (const float4*)xl, aex, denom, ei, hout, C4);

    h = hout;
    hout = (hout == h0) ? h1 : h0;
  }

  // ---- global mean pool (fused relu) ----
  int poolElems = N_GRAPHS * NHID + N_GRAPHS;        // pools and cnt adjacent
  zero_kernel<<<(poolElems + 255) / 256, 256, 0, stream>>>(pools, poolElems);
  pool_scatter_kernel<<<(N_NODES + 7) / 8, 256, 0, stream>>>(
      (const float4*)h, batch, pools, cnt, NHID / 4);
  pool_div_kernel<<<(N_GRAPHS * NHID + 255) / 256, 256, 0, stream>>>(pools, cnt, gbuf, NHID);

  // ---- FC head (both via WMMA) ----
  const float* fc1_W = (const float*)d_in[27];
  const float* fc1_b = (const float*)d_in[28];
  const float* fc2_W = (const float*)d_in[29];
  const float* fc2_b = (const float*)d_in[30];
  const int KpadF = ceil32(NHID);                    // 320

  launch_convert_a(gbuf, gh, N_GRAPHS, NHID, KpadF, 0, stream);
  launch_convert_w(fc1_W, wtA, NHID, NHID, KpadF, ((NHID + 63) / 64) * 64, stream);
  launch_gemm(gh, wtA, fc1_b, fc1o, N_GRAPHS, NHID, KpadF, 1, stream);

  launch_convert_a(fc1o, gh, N_GRAPHS, NHID, KpadF, 0, stream);
  launch_convert_w(fc2_W, wtB, NHID, NOUT, KpadF, ((NOUT + 63) / 64) * 64, stream);
  launch_gemm(gh, wtB, fc2_b, (float*)d_out, N_GRAPHS, NOUT, KpadF, 0, stream);
}
